// CodecT_46591805227620
// MI455X (gfx1250) — compile-verified, matrix operand order checked by reading, not run on hardware
//
#include <hip/hip_runtime.h>
#include <hip/hip_bf16.h>

// ---------------------------------------------------------------------------
// CDNA5 (gfx1250) implementation of the VQ transformer codec.
// All GEMMs (linear stacks, attention projections, QK^T, AV, FFN) run through
// one WMMA kernel: V_WMMA_F32_16X16X32_F16, f32 accumulate, fused
// bias/activation/residual epilogue. LN / softmax / conv / VQ are VALU kernels.
// ---------------------------------------------------------------------------

typedef __attribute__((ext_vector_type(16))) _Float16 v16h;
typedef __attribute__((ext_vector_type(8)))  float    v8f;

static constexpr int  Bn  = 8;
static constexpr int  Tn  = 512;
static constexpr int  Dm  = 768;          // model dim
static constexpr int  Hn  = 4;            // heads
static constexpr int  HD  = Dm / Hn;      // 192
static constexpr int  BT  = Bn * Tn;      // 4096 rows
static constexpr int  DIN = 80;
static constexpr int  DP  = 64;
static constexpr int  DL  = 256;
static constexpr long long S = (long long)BT * Dm;   // elements of a (B,T,D) tensor
static constexpr float ATT_SCALE = 0.07216878364870323f; // 1/sqrt(192)
static constexpr int  LDP = 40;           // LDS row pitch in halves (80B = 16B multiple)

__device__ __forceinline__ int imin(int a, int b) { return a < b ? a : b; }

union FragU { uint4 q[2]; v16h h; };      // two 16B chunks -> one WMMA fragment
union PackU { _Float16 h[8]; uint4 q; };  // 8 packed halves -> one b128 store

// ---------------------------------------------------------------------------
// WMMA GEMM:  C[M,N] = act(alpha * A[M,K] x B + bias[N]) + res[M,N]
//   TRANSB=1 -> B stored [N,K] (i.e. W row-major, used as W^T)
//   TRANSB=0 -> B stored [K,N]
//   Batched via gridDim.z: offset = (z/bh)*s1 + (z%bh)*s2 for each operand.
// Block: 256 threads = 8 waves. Block tile 64x64, K step 32.
// Wave w: rows (w&3)*16..+15, cols (w>>2)*32..+31  (two 16x16 WMMA tiles).
// LDS rows padded to 40 halves (80B) so every 8-half group is 16B-aligned:
// staging writes are single ds_store_b128, fragment gathers are ds_load_b128.
// K loop is versioned: branch-free main loop over the 32-aligned part of K,
// single guarded tail tile after it.
// ---------------------------------------------------------------------------
template <int TRANSB>
__global__ __launch_bounds__(256)
void k_gemm(const float* __restrict__ A, long long sA1, long long sA2, int lda,
            const float* __restrict__ Bm, long long sB1, long long sB2, int ldb,
            const float* __restrict__ bias,
            const float* __restrict__ res, long long sR1, long long sR2, int ldr,
            float* __restrict__ C, long long sC1, long long sC2, int ldc,
            int M, int N, int K, int bh, int act, float alpha)
{
    __shared__ alignas(16) _Float16 As[64][LDP];   // [m][k]
    __shared__ alignas(16) _Float16 Bs[64][LDP];   // [n][k]

    const int z  = blockIdx.z;
    const int zb = z / bh, zh = z % bh;
    A  += zb * sA1 + zh * sA2;
    Bm += zb * sB1 + zh * sB2;
    if (res) res += zb * sR1 + zh * sR2;
    C  += zb * sC1 + zh * sC2;

    const int m0  = blockIdx.y * 64;
    const int n0  = blockIdx.x * 64;
    const int tid = threadIdx.x;
    const int lane = tid & 31;
    const int wv   = tid >> 5;
    const int wm   = (wv & 3) * 16;   // wave row offset inside tile
    const int wn   = (wv >> 2) * 32;  // wave col offset inside tile

    v8f acc0 = {};
    v8f acc1 = {};

    const int lrow = tid >> 2;        // 0..63 (A row / B col owned by this thread)
    const int lk0  = (tid & 3) * 8;   // 0,8,16,24
    const int krow = tid >> 3;        // 0..31 (TRANSB=0 staging)
    const int nc0  = (tid & 7) * 8;   // 0..56 (TRANSB=0 staging)

    const bool aRowIn = (m0 + 64 <= M);          // whole A tile rows in-bounds
    const bool bColIn = (n0 + 64 <= N);          // whole B tile cols in-bounds

    // ---- stage one 64x32 (A) + 64x32 (B) tile pair at k offset kt ----
    auto stage = [&](int kt, bool fullK) {
        // A tile (f32 -> f16)
        {
            const int gr = m0 + lrow;
            if (aRowIn && fullK) {
                const float* ap = A + (long long)gr * lda + kt + lk0;
                const float4 f0 = *(const float4*)(ap);
                const float4 f1 = *(const float4*)(ap + 4);
                PackU p;
                p.h[0] = (_Float16)f0.x; p.h[1] = (_Float16)f0.y;
                p.h[2] = (_Float16)f0.z; p.h[3] = (_Float16)f0.w;
                p.h[4] = (_Float16)f1.x; p.h[5] = (_Float16)f1.y;
                p.h[6] = (_Float16)f1.z; p.h[7] = (_Float16)f1.w;
                *(uint4*)&As[lrow][lk0] = p.q;
            } else {
                const int cr = imin(gr, M - 1);
                _Float16* dst = &As[lrow][lk0];
                #pragma unroll
                for (int i = 0; i < 8; ++i) {
                    const int gk = kt + lk0 + i;
                    const float v = A[(long long)cr * lda + imin(gk, K - 1)];
                    dst[i] = (gr < M && gk < K) ? (_Float16)v : (_Float16)0.0f;
                }
            }
        }
        // B tile into Bs[n][k]
        if (TRANSB) {
            const int gn = n0 + lrow;
            if (bColIn && fullK) {
                const float* bp = Bm + (long long)gn * ldb + kt + lk0;
                const float4 f0 = *(const float4*)(bp);
                const float4 f1 = *(const float4*)(bp + 4);
                PackU p;
                p.h[0] = (_Float16)f0.x; p.h[1] = (_Float16)f0.y;
                p.h[2] = (_Float16)f0.z; p.h[3] = (_Float16)f0.w;
                p.h[4] = (_Float16)f1.x; p.h[5] = (_Float16)f1.y;
                p.h[6] = (_Float16)f1.z; p.h[7] = (_Float16)f1.w;
                *(uint4*)&Bs[lrow][lk0] = p.q;
            } else {
                const int cn = imin(gn, N - 1);
                _Float16* dst = &Bs[lrow][lk0];
                #pragma unroll
                for (int i = 0; i < 8; ++i) {
                    const int gk = kt + lk0 + i;
                    const float v = Bm[(long long)cn * ldb + imin(gk, K - 1)];
                    dst[i] = (gn < N && gk < K) ? (_Float16)v : (_Float16)0.0f;
                }
            }
        } else {
            // B stored [K,N]: stream float4 along N at fixed k row, scatter to [n][k]
            const int gk = kt + krow;
            if (bColIn && fullK) {       // TRANSB=0 users are K%32==0, gk always valid
                const float* bp = Bm + (long long)gk * ldb + n0 + nc0;
                const float4 f0 = *(const float4*)(bp);
                const float4 f1 = *(const float4*)(bp + 4);
                Bs[nc0 + 0][krow] = (_Float16)f0.x;
                Bs[nc0 + 1][krow] = (_Float16)f0.y;
                Bs[nc0 + 2][krow] = (_Float16)f0.z;
                Bs[nc0 + 3][krow] = (_Float16)f0.w;
                Bs[nc0 + 4][krow] = (_Float16)f1.x;
                Bs[nc0 + 5][krow] = (_Float16)f1.y;
                Bs[nc0 + 6][krow] = (_Float16)f1.z;
                Bs[nc0 + 7][krow] = (_Float16)f1.w;
            } else {
                const int ck = imin(gk, K - 1);
                #pragma unroll
                for (int i = 0; i < 8; ++i) {
                    const int gn = n0 + nc0 + i;
                    const float v = Bm[(long long)ck * ldb + imin(gn, N - 1)];
                    Bs[nc0 + i][krow] = (gn < N && gk < K) ? (_Float16)v : (_Float16)0.0f;
                }
            }
        }
    };

    // ---- fragment build + 2 WMMA ----
    // ISA 16-bit fragment layout: lane L holds, for M/N index (L&15), the
    // contiguous k-groups [hi*8, hi*8+8) and [16+hi*8, 16+hi*8+8) (hi = L>>4).
    // Both groups are 16B-aligned in LDS -> two ds_load_b128 per fragment.
    auto compute = [&]() {
        const int l16 = lane & 15;
        const int hi  = lane >> 4;
        const _Float16* ar = &As[wm + l16][hi * 8];
        const _Float16* b0 = &Bs[wn + l16][hi * 8];
        const _Float16* b1 = &Bs[wn + 16 + l16][hi * 8];
        FragU fa, fb0, fb1;
        fa.q[0]  = *(const uint4*)(ar);
        fa.q[1]  = *(const uint4*)(ar + 16);
        fb0.q[0] = *(const uint4*)(b0);
        fb0.q[1] = *(const uint4*)(b0 + 16);
        fb1.q[0] = *(const uint4*)(b1);
        fb1.q[1] = *(const uint4*)(b1 + 16);
        acc0 = __builtin_amdgcn_wmma_f32_16x16x32_f16(false, fa.h, false, fb0.h,
                                                      (short)0, acc0, false, false);
        acc1 = __builtin_amdgcn_wmma_f32_16x16x32_f16(false, fa.h, false, fb1.h,
                                                      (short)0, acc1, false, false);
    };

    const int Kmain = K & ~31;
    for (int kt = 0; kt < Kmain; kt += 32) {
        stage(kt, true);                 // fullK constant-folds; loop unswitches on
        __syncthreads();                 // the loop-invariant aRowIn/bColIn
        compute();
        __syncthreads();
    }
    if (Kmain < K) {                     // single K-remainder tile (K=80 stacks only)
        stage(Kmain, false);
        __syncthreads();
        compute();
        __syncthreads();
    }

    // ---- epilogue: alpha, bias, activation, residual, store ----
    const int l16 = lane & 15;
    const int hi  = lane >> 4;
    #pragma unroll
    for (int r = 0; r < 8; ++r) {
        const int row = m0 + wm + hi * 8 + r;
        if (row >= M) continue;
        int col = n0 + wn + l16;
        if (col < N) {
            float y = acc0[r] * alpha + (bias ? bias[col] : 0.0f);
            if (act == 1) y = fmaxf(y, 0.0f);
            else if (act == 2) y = (y > 0.0f) ? y : 0.01f * y;
            if (res) y += res[(long long)row * ldr + col];
            C[(long long)row * ldc + col] = y;
        }
        col = n0 + wn + 16 + l16;
        if (col < N) {
            float y = acc1[r] * alpha + (bias ? bias[col] : 0.0f);
            if (act == 1) y = fmaxf(y, 0.0f);
            else if (act == 2) y = (y > 0.0f) ? y : 0.01f * y;
            if (res) y += res[(long long)row * ldr + col];
            C[(long long)row * ldc + col] = y;
        }
    }
}

// ---------------------------------------------------------------------------
// Row LayerNorm over D elements (in-place safe).
// ---------------------------------------------------------------------------
__global__ __launch_bounds__(256)
void k_layernorm(const float* __restrict__ x, const float* __restrict__ g,
                 const float* __restrict__ b, float* __restrict__ y, int D)
{
    const long long row = blockIdx.x;
    const float* xr = x + row * D;
    float* yr = y + row * D;
    __shared__ float red[256];

    float s = 0.0f;
    for (int i = threadIdx.x; i < D; i += 256) s += xr[i];
    red[threadIdx.x] = s; __syncthreads();
    for (int o = 128; o > 0; o >>= 1) {
        if (threadIdx.x < o) red[threadIdx.x] += red[threadIdx.x + o];
        __syncthreads();
    }
    const float mean = red[0] / D;
    __syncthreads();

    s = 0.0f;
    for (int i = threadIdx.x; i < D; i += 256) { float d = xr[i] - mean; s += d * d; }
    red[threadIdx.x] = s; __syncthreads();
    for (int o = 128; o > 0; o >>= 1) {
        if (threadIdx.x < o) red[threadIdx.x] += red[threadIdx.x + o];
        __syncthreads();
    }
    const float rstd = rsqrtf(red[0] / D + 1e-5f);
    for (int i = threadIdx.x; i < D; i += 256)
        yr[i] = g[i] * (xr[i] - mean) * rstd + b[i];
}

// ---------------------------------------------------------------------------
// Row softmax with optional causal mask (scores already scaled in GEMM).
// grid: (T rows, B*H batches)
// ---------------------------------------------------------------------------
__global__ __launch_bounds__(256)
void k_softmax(float* __restrict__ Sc, int T, int causal)
{
    const int q = blockIdx.x;
    float* row = Sc + ((long long)blockIdx.y * T + q) * T;
    const int lim = causal ? (q + 1) : T;
    __shared__ float red[256];

    float mx = -1e30f;
    for (int i = threadIdx.x; i < T; i += 256) {
        float v = (i < lim) ? row[i] : -1e9f;
        mx = fmaxf(mx, v);
    }
    red[threadIdx.x] = mx; __syncthreads();
    for (int o = 128; o > 0; o >>= 1) {
        if (threadIdx.x < o) red[threadIdx.x] = fmaxf(red[threadIdx.x], red[threadIdx.x + o]);
        __syncthreads();
    }
    mx = red[0];
    __syncthreads();

    float sum = 0.0f;
    for (int i = threadIdx.x; i < T; i += 256) {
        float v = (i < lim) ? row[i] : -1e9f;
        sum += __expf(v - mx);
    }
    red[threadIdx.x] = sum; __syncthreads();
    for (int o = 128; o > 0; o >>= 1) {
        if (threadIdx.x < o) red[threadIdx.x] += red[threadIdx.x + o];
        __syncthreads();
    }
    const float inv = 1.0f / red[0];
    for (int i = threadIdx.x; i < T; i += 256) {
        float v = (i < lim) ? row[i] : -1e9f;
        row[i] = __expf(v - mx) * inv;
    }
}

// ---------------------------------------------------------------------------
// Conv1d(C=D, O=D, k=4, stride=2, pad=1). x: (B,T,D) frame-major,
// y: (B,T/2,D) frame-major.  w: (O,C,4).  Optional fused LeakyReLU.
// ---------------------------------------------------------------------------
__global__ __launch_bounds__(256)
void k_conv1d_s2(const float* __restrict__ x, const float* __restrict__ w,
                 const float* __restrict__ bias, float* __restrict__ y,
                 int B_, int T, int D, int leaky)
{
    const long long i = (long long)blockIdx.x * 256 + threadIdx.x;
    const int T2 = T >> 1;
    const long long total = (long long)B_ * T2 * D;
    if (i >= total) return;
    const int o  = (int)(i % D);
    const long long f = i / D;
    const int t2 = (int)(f % T2);
    const int b  = (int)(f / T2);

    float acc = bias[o];
    const float* wr = w + (long long)o * D * 4;
    #pragma unroll
    for (int k = 0; k < 4; ++k) {
        const int tt = 2 * t2 - 1 + k;
        if (tt < 0 || tt >= T) continue;
        const float* xr = x + ((long long)b * T + tt) * D;
        for (int c = 0; c < D; ++c) acc += wr[c * 4 + k] * xr[c];
    }
    if (leaky) acc = (acc > 0.0f) ? acc : 0.01f * acc;
    y[i] = acc;
}

// out[b,t,d] = in[b, t/rate, d]   (time upsample by repetition)
__global__ __launch_bounds__(256)
void k_repeat(const float* __restrict__ in, float* __restrict__ out,
              long long total, int T, int D, int rate)
{
    const long long i = (long long)blockIdx.x * 256 + threadIdx.x;
    if (i >= total) return;
    const int d = (int)(i % D);
    const long long f = i / D;
    const int t = (int)(f % T);
    const int b = (int)(f / T);
    out[i] = in[((long long)b * (T / rate) + t / rate) * D + d];
}

__global__ __launch_bounds__(256)
void k_sub(const float* __restrict__ a, const float* __restrict__ b,
           float* __restrict__ o, long long n)
{
    const long long i = (long long)blockIdx.x * 256 + threadIdx.x;
    if (i < n) o[i] = a[i] - b[i];
}

__global__ __launch_bounds__(256)
void k_add(const float* __restrict__ a, const float* __restrict__ b,
           float* __restrict__ o, long long n)
{
    const long long i = (long long)blockIdx.x * 256 + threadIdx.x;
    if (i < n) o[i] = a[i] + b[i];
}

// ---------------------------------------------------------------------------
// VQ: one block (256 threads) per row; thread j owns codeword j.
// q[row] = cb[argmin_j ||x[row]-cb[j]||^2]
// ---------------------------------------------------------------------------
__global__ __launch_bounds__(256)
void k_vq(const float* __restrict__ x, const float* __restrict__ cb,
          float* __restrict__ q, int D)
{
    const long long row = blockIdx.x;
    __shared__ float xs[Dm];
    __shared__ float db[256];
    __shared__ int   ib[256];
    const int t = threadIdx.x;

    for (int i = t; i < D; i += 256) xs[i] = x[row * D + i];
    __syncthreads();

    const float* cw = cb + (long long)t * D;
    float d = 0.0f;
    for (int i = 0; i < D; ++i) { float df = xs[i] - cw[i]; d += df * df; }
    db[t] = d; ib[t] = t; __syncthreads();
    for (int o = 128; o > 0; o >>= 1) {
        if (t < o) {
            if (db[t + o] < db[t] || (db[t + o] == db[t] && ib[t + o] < ib[t])) {
                db[t] = db[t + o]; ib[t] = ib[t + o];
            }
        }
        __syncthreads();
    }
    const int best = ib[0];
    for (int i = t; i < D; i += 256) q[row * D + i] = cb[(long long)best * D + i];
}

// ===========================================================================
// Host-side orchestration
// ===========================================================================
struct LNW  { const float *g, *b; };
struct AttW { const float *wq,*bq,*wk,*bk,*wv,*bv,*wo,*bo; };
struct FfnW { const float *w1,*b1,*w2,*b2; };
struct EncL { LNW ln1; AttW at; LNW ln2; FfnW ff; };
struct DecL { LNW ln1; AttW sa; LNW ln2; AttW ca; LNW ln3; FfnW ff; };
struct EncBlk { EncL l[4]; LNW norm; const float* dsw[2]; const float* dsb[2]; int nds; };
struct DecBlk { DecL l[4]; LNW norm; };
struct Stack  { const float* w[4]; const float* b[4]; int n; };
struct Bufs   { float *h, *ff, *qb, *kb, *vb, *sc, *ob; };

static inline const float* P(void* const* din, int& t) { return (const float*)din[t++]; }
static LNW  parseLN (void* const* d, int& t){ LNW x; x.g=P(d,t); x.b=P(d,t); return x; }
static AttW parseAtt(void* const* d, int& t){ AttW a; a.wq=P(d,t); a.bq=P(d,t); a.wk=P(d,t); a.bk=P(d,t);
                                              a.wv=P(d,t); a.bv=P(d,t); a.wo=P(d,t); a.bo=P(d,t); return a; }
static FfnW parseFfn(void* const* d, int& t){ FfnW f; f.w1=P(d,t); f.b1=P(d,t); f.w2=P(d,t); f.b2=P(d,t); return f; }
static EncL parseEncL(void* const* d, int& t){ EncL e; e.ln1=parseLN(d,t); e.at=parseAtt(d,t);
                                               e.ln2=parseLN(d,t); e.ff=parseFfn(d,t); return e; }
static DecL parseDecL(void* const* d, int& t){ DecL e; e.ln1=parseLN(d,t); e.sa=parseAtt(d,t);
                                               e.ln2=parseLN(d,t); e.ca=parseAtt(d,t);
                                               e.ln3=parseLN(d,t); e.ff=parseFfn(d,t); return e; }
static Stack parseStack(void* const* d, int& t, int n){
    Stack s; s.n = n;
    for (int i = 0; i < n; ++i) s.w[i] = P(d, t);
    for (int i = 0; i < n; ++i) s.b[i] = P(d, t);
    return s;
}

static inline void launch_gemm(hipStream_t st,
    const float* A, long long sA1, long long sA2, int lda,
    const float* Bm, long long sB1, long long sB2, int ldb, int transB,
    const float* bias,
    const float* res, long long sR1, long long sR2, int ldr,
    float* C, long long sC1, long long sC2, int ldc,
    int M, int N, int K, int nb, int bh, int act, float alpha)
{
    dim3 grid((N + 63) / 64, (M + 63) / 64, nb);
    if (transB)
        k_gemm<1><<<grid, 256, 0, st>>>(A, sA1, sA2, lda, Bm, sB1, sB2, ldb,
                                        bias, res, sR1, sR2, ldr, C, sC1, sC2, ldc,
                                        M, N, K, bh, act, alpha);
    else
        k_gemm<0><<<grid, 256, 0, st>>>(A, sA1, sA2, lda, Bm, sB1, sB2, ldb,
                                        bias, res, sR1, sR2, ldr, C, sC1, sC2, ldc,
                                        M, N, K, bh, act, alpha);
}

// Unbatched linear: C = act(A @ W^T + bias) [+ res], W stored [N,K]
static inline void linear(hipStream_t st, const float* A, int lda, const float* W,
                          int K, int N, const float* bias, const float* res,
                          float* C, int ldc, int M, int act)
{
    launch_gemm(st, A, 0, 0, lda, W, 0, 0, K, 1, bias, res, 0, 0, ldc,
                C, 0, 0, ldc, M, N, K, 1, 1, act, 1.0f);
}

static inline void layernorm(hipStream_t st, const float* x, const LNW& ln, float* y)
{
    k_layernorm<<<BT, 256, 0, st>>>(x, ln.g, ln.b, y, Dm);
}

static void attention(hipStream_t st, const float* qin, const float* kvin,
                      const AttW& w, bool causal, const float* residual,
                      float* out, const Bufs& bf)
{
    linear(st, qin,  Dm, w.wq, Dm, Dm, w.bq, nullptr, bf.qb, Dm, BT, 0);
    linear(st, kvin, Dm, w.wk, Dm, Dm, w.bk, nullptr, bf.kb, Dm, BT, 0);
    linear(st, kvin, Dm, w.wv, Dm, Dm, w.bv, nullptr, bf.vb, Dm, BT, 0);
    // scores[b,h] = (Q_head @ K_head^T) / sqrt(hd)
    launch_gemm(st, bf.qb, (long long)Tn * Dm, HD, Dm,
                    bf.kb, (long long)Tn * Dm, HD, Dm, 1,
                    nullptr, nullptr, 0, 0, 0,
                    bf.sc, (long long)Hn * Tn * Tn, (long long)Tn * Tn, Tn,
                    Tn, Tn, HD, Bn * Hn, Hn, 0, ATT_SCALE);
    k_softmax<<<dim3(Tn, Bn * Hn), 256, 0, st>>>(bf.sc, Tn, causal ? 1 : 0);
    // O_head = A @ V_head
    launch_gemm(st, bf.sc, (long long)Hn * Tn * Tn, (long long)Tn * Tn, Tn,
                    bf.vb, (long long)Tn * Dm, HD, Dm, 0,
                    nullptr, nullptr, 0, 0, 0,
                    bf.ob, (long long)Tn * Dm, HD, Dm,
                    Tn, HD, Tn, Bn * Hn, Hn, 0, 1.0f);
    linear(st, bf.ob, Dm, w.wo, Dm, Dm, w.bo, residual, out, Dm, BT, 0);
}

static void ffn(hipStream_t st, const float* h, const FfnW& f, float* x, const Bufs& bf)
{
    linear(st, h, Dm, f.w1, Dm, Dm, f.b1, nullptr, bf.ff, Dm, BT, 1 /*relu*/);
    linear(st, bf.ff, Dm, f.w2, Dm, Dm, f.b2, x, x, Dm, BT, 0);
}

static void enc_block(hipStream_t st, const float* in, float* out,
                      const EncBlk& bk, const Bufs& bf)
{
    hipMemcpyAsync(out, in, S * sizeof(float), hipMemcpyDeviceToDevice, st);
    for (int i = 0; i < 4; ++i) {
        const EncL& L = bk.l[i];
        layernorm(st, out, L.ln1, bf.h);
        attention(st, bf.h, bf.h, L.at, false, out, out, bf);
        layernorm(st, out, L.ln2, bf.h);
        ffn(st, bf.h, L.ff, out, bf);
    }
    layernorm(st, out, bk.norm, out);
}

static void dec_block(hipStream_t st, const float* in, const float* mem,
                      float* out, const DecBlk& bk, const Bufs& bf)
{
    hipMemcpyAsync(out, in, S * sizeof(float), hipMemcpyDeviceToDevice, st);
    for (int i = 0; i < 4; ++i) {
        const DecL& L = bk.l[i];
        layernorm(st, out, L.ln1, bf.h);
        attention(st, bf.h, bf.h, L.sa, true, out, out, bf);
        layernorm(st, out, L.ln2, bf.h);
        attention(st, bf.h, mem, L.ca, false, out, out, bf);
        layernorm(st, out, L.ln3, bf.h);
        ffn(st, bf.h, L.ff, out, bf);
    }
    layernorm(st, out, bk.norm, out);
}

// per-frame 1x1 conv stack; LeakyReLU between layers, last layer linear
static void stack_apply(hipStream_t st, const Stack& s, const int* dims,
                        const float* in, float* out_final, int ld_final,
                        float* tA, float* tB)
{
    const float* cur = in;
    int curld = dims[0];
    for (int i = 0; i < s.n; ++i) {
        const bool last = (i == s.n - 1);
        float* dst = last ? out_final : ((i & 1) ? tB : tA);
        const int ldd = last ? ld_final : dims[i + 1];
        linear(st, cur, curld, s.w[i], dims[i], dims[i + 1], s.b[i], nullptr,
               dst, ldd, BT, last ? 0 : 2 /*leaky*/);
        cur = dst; curld = ldd;
    }
}

static inline long long gblocks(long long n) { return (n + 255) / 256; }

extern "C" void kernel_launch(void* const* d_in, const int* in_sizes, int n_in,
                              void* d_out, int out_size, void* d_ws, size_t ws_size,
                              hipStream_t stream)
{
    (void)in_sizes; (void)n_in; (void)out_size; (void)ws_size;
    // ---- unpack inputs (depth-first, dict-insertion order; 448 leaves) ----
    int t = 0;
    const float* x     = P(d_in, t);
    const float* pitch = P(d_in, t);
    const float* mag   = P(d_in, t);
    Stack enc_lin   = parseStack(d_in, t, 3);
    Stack pitch_lin = parseStack(d_in, t, 3);
    Stack mag_lin   = parseStack(d_in, t, 3);

    EncBlk blk1, blk2, blk3;
    for (int i = 0; i < 4; ++i) blk1.l[i] = parseEncL(d_in, t);
    blk1.norm = parseLN(d_in, t);
    blk1.dsw[0] = P(d_in, t); blk1.dsb[0] = P(d_in, t);
    blk1.dsw[1] = P(d_in, t); blk1.dsb[1] = P(d_in, t); blk1.nds = 2;
    for (int i = 0; i < 4; ++i) blk2.l[i] = parseEncL(d_in, t);
    blk2.norm = parseLN(d_in, t);
    blk2.dsw[0] = P(d_in, t); blk2.dsb[0] = P(d_in, t); blk2.nds = 1;
    for (int i = 0; i < 4; ++i) blk3.l[i] = parseEncL(d_in, t);
    blk3.norm = parseLN(d_in, t); blk3.nds = 0;

    const float* cb1 = P(d_in, t);
    const float* cb2 = P(d_in, t);
    const float* cb3 = P(d_in, t);

    DecBlk dec2, dec3;
    for (int i = 0; i < 4; ++i) dec2.l[i] = parseDecL(d_in, t);
    dec2.norm = parseLN(d_in, t);
    for (int i = 0; i < 4; ++i) dec3.l[i] = parseDecL(d_in, t);
    dec3.norm = parseLN(d_in, t);
    Stack dec_lin = parseStack(d_in, t, 4);

    // ---- workspace bump allocator (requires ~280 MB) ----
    float* ws = (float*)d_ws;
    size_t off = 0;
    auto alloc = [&](long long n) { float* p = ws + off; off += (size_t)n; return p; };

    float* cbuf = alloc(S);            // concat(xe,pe,me)
    float* e1   = alloc(S);            // enc block out (x1/x2/x3 reuse)
    float* dsb  = alloc(S);            // upsampled ds (ds1/ds2 reuse)
    float* q1   = alloc(S);
    float* q2   = alloc(S);
    float* q3   = alloc(S);
    float* r1   = alloc(S);
    float* r2   = alloc(S);
    float* d1   = alloc(S);
    float* d2   = alloc(S);
    float* d3   = alloc(S);
    float* tsum = alloc(S);
    Bufs bf;
    bf.h  = alloc(S);
    bf.ff = alloc(S);
    bf.qb = alloc(S);
    bf.kb = alloc(S);
    bf.vb = alloc(S);
    bf.ob = alloc(S);
    bf.sc = alloc((long long)Bn * Hn * Tn * Tn);
    float* convA = alloc(S / 2);       // (B, T/2, D)
    float* convB = alloc(S / 4);       // (B, T/4, D)
    float* t256a = alloc((long long)BT * DL);
    float* t256b = alloc((long long)BT * DL);

    // ---- input 1x1-conv stacks -> concat into cbuf columns [0:256|256:512|512:768]
    { const int dims[4] = {DIN, DL, DL, DL};
      stack_apply(stream, enc_lin,   dims, x,     cbuf + 0,      Dm, t256a, t256b); }
    { const int dims[4] = {DP, DL, DL, DL};
      stack_apply(stream, pitch_lin, dims, pitch, cbuf + DL,     Dm, t256a, t256b); }
    { const int dims[4] = {DP, DL, DL, DL};
      stack_apply(stream, mag_lin,   dims, mag,   cbuf + 2 * DL, Dm, t256a, t256b); }

    // ---- encoder stage 1 (rate 4) ----
    enc_block(stream, cbuf, e1, blk1, bf);                               // x1 = e1
    k_conv1d_s2<<<gblocks((long long)Bn * (Tn/2) * Dm), 256, 0, stream>>>(
        e1, blk1.dsw[0], blk1.dsb[0], convA, Bn, Tn, Dm, 1 /*leaky*/);
    k_conv1d_s2<<<gblocks((long long)Bn * (Tn/4) * Dm), 256, 0, stream>>>(
        convA, blk1.dsw[1], blk1.dsb[1], convB, Bn, Tn / 2, Dm, 0);
    k_repeat<<<gblocks(S), 256, 0, stream>>>(convB, dsb, S, Tn, Dm, 4);  // ds1
    k_vq<<<BT, 256, 0, stream>>>(dsb, cb1, q1, Dm);
    k_sub<<<gblocks(S), 256, 0, stream>>>(e1, q1, r1, S);                // r1 = x1 - q1

    // ---- encoder stage 2 (rate 2) ----
    enc_block(stream, r1, e1, blk2, bf);                                 // x2 = e1
    k_conv1d_s2<<<gblocks((long long)Bn * (Tn/2) * Dm), 256, 0, stream>>>(
        e1, blk2.dsw[0], blk2.dsb[0], convA, Bn, Tn, Dm, 0);
    k_repeat<<<gblocks(S), 256, 0, stream>>>(convA, dsb, S, Tn, Dm, 2);  // ds2
    k_vq<<<BT, 256, 0, stream>>>(dsb, cb2, q2, Dm);
    k_sub<<<gblocks(S), 256, 0, stream>>>(e1, q2, r2, S);                // r2 = x2 - q2

    // ---- encoder stage 3 (rate 1) ----
    enc_block(stream, r2, e1, blk3, bf);                                 // x3 = e1
    k_vq<<<BT, 256, 0, stream>>>(e1, cb3, q3, Dm);

    // ---- decoders ----
    dec_block(stream, cbuf, q1, d1, dec3, bf);                           // d1
    dec_block(stream, r1,   q2, d2, dec2, bf);
    k_add<<<gblocks(S), 256, 0, stream>>>(d2, d1, tsum, S);
    dec_block(stream, cbuf, tsum, d2, dec2, bf);                         // d2
    dec_block(stream, r2,   q3, d3, dec2, bf);
    k_add<<<gblocks(S), 256, 0, stream>>>(d3, d2, tsum, S);
    dec_block(stream, r1,   tsum, d3, dec2, bf);
    k_add<<<gblocks(S), 256, 0, stream>>>(d3, d1, tsum, S);
    dec_block(stream, cbuf, tsum, d3, dec2, bf);                         // d3

    // ---- output stacks -> d_out[3, B, T, 80] ----
    float* out = (float*)d_out;
    const int odims[5] = {Dm, DL, DL, DL, DIN};
    stack_apply(stream, dec_lin, odims, d1, out + 0LL * BT * DIN, DIN, t256a, t256b);
    stack_apply(stream, dec_lin, odims, d2, out + 1LL * BT * DIN, DIN, t256a, t256b);
    stack_apply(stream, dec_lin, odims, d3, out + 2LL * BT * DIN, DIN, t256a, t256b);
}
